// GraphTransformerDenoiser_35416300322908
// MI455X (gfx1250) — matrix operand order repaired and checked by reading, not compiled, not run on hardware
//
#include <hip/hip_runtime.h>
#include <hip/hip_bf16.h>

// ---------------------------------------------------------------------------
// GraphTransformerDenoiser for MI455X (gfx1250).
// Dense GEMMs -> v_wmma_f32_16x16x32_f16 (f16 in / f32 acc), register-blocked
// NT N-tiles per wave for A reuse.  Scatter attention -> streaming VALU +
// global float atomics (memory bound at 23.3 TB/s).
// ---------------------------------------------------------------------------

#define NN    16000
#define EE    256000
#define HH    192
#define HEADS 4
#define CC    48
#define LL    4
#define BB    8
#define VN_   32
#define VE_   8

typedef __attribute__((ext_vector_type(16))) _Float16 v16h;
typedef __attribute__((ext_vector_type(8)))  _Float16 v8h;
typedef __attribute__((ext_vector_type(8)))  float    v8f;

// ------------------------- small helpers -----------------------------------

__device__ __forceinline__ unsigned fmap(float f) {
  unsigned u = __float_as_uint(f);
  return (u & 0x80000000u) ? ~u : (u | 0x80000000u);
}
__device__ __forceinline__ float funmap(unsigned u) {
  return (u & 0x80000000u) ? __uint_as_float(u & 0x7fffffffu)
                           : __uint_as_float(~u);
}
__device__ __forceinline__ float silu_f(float x) {
  return x / (1.f + __expf(-x));
}

// ------------------- weight convert + transpose (f32 -> f16^T) -------------
// W[K,Nc] (row major f32)  ->  Wt[Nc,K] (row major f16)
__global__ void gtd_convT(const float* __restrict__ w, _Float16* __restrict__ wt,
                          int K, int Nc) {
  int t = blockIdx.x * blockDim.x + threadIdx.x;
  if (t >= K * Nc) return;
  int kk = t / Nc, n = t % Nc;
  wt[(size_t)n * K + kk] = (_Float16)w[t];
}

// ------------------------- time embedding ----------------------------------
// one block per graph b (B=8), 192 threads
__global__ void gtd_time_embed(const int* __restrict__ tg,
                               const float* __restrict__ w1, const float* __restrict__ b1,
                               const float* __restrict__ w2, const float* __restrict__ b2,
                               float* __restrict__ tvec, _Float16* __restrict__ tvech) {
  __shared__ float sbuf[HH];
  __shared__ float hbuf[HH];
  int b = blockIdx.x, j = threadIdx.x;
  float t = (float)tg[b];
  float val;
  if (j < 96) {
    float f = __expf(-logf(10000.f) * (float)j / 96.f);
    val = __sinf(t * f);
  } else {
    float f = __expf(-logf(10000.f) * (float)(j - 96) / 96.f);
    val = __cosf(t * f);
  }
  sbuf[j] = val;
  __syncthreads();
  float acc = b1[j];
  for (int i = 0; i < HH; ++i) acc += sbuf[i] * w1[i * HH + j];
  acc = silu_f(acc);
  hbuf[j] = acc;
  __syncthreads();
  float acc2 = b2[j];
  for (int i = 0; i < HH; ++i) acc2 += hbuf[i] * w2[i * HH + j];
  tvec[b * HH + j]  = acc2;
  tvech[b * HH + j] = (_Float16)acc2;
}

// ------------------------- input embedding gathers -------------------------
__global__ void gtd_init_x(const float* __restrict__ node_emb,
                           const float* __restrict__ mask_emb,
                           const float* __restrict__ tvec,
                           const int* __restrict__ node_type,
                           const int* __restrict__ edit_mask,
                           const int* __restrict__ node_batch,
                           float* __restrict__ x, _Float16* __restrict__ xh) {
  long long t = (long long)blockIdx.x * blockDim.x + threadIdx.x;
  if (t >= (long long)NN * HH) return;
  int n = (int)(t / HH), c = (int)(t % HH);
  float v = node_emb[node_type[n] * HH + c] + tvec[node_batch[n] * HH + c] +
            mask_emb[edit_mask[n] * HH + c];
  x[t] = v; xh[t] = (_Float16)v;
}

__global__ void gtd_init_edge(const float* __restrict__ edge_emb,
                              const float* __restrict__ tvec,
                              const int* __restrict__ edge_type,
                              const int* __restrict__ srcI,
                              const int* __restrict__ node_batch,
                              float* __restrict__ ea, _Float16* __restrict__ eah) {
  long long t = (long long)blockIdx.x * blockDim.x + threadIdx.x;
  if (t >= (long long)EE * HH) return;
  int e = (int)(t / HH), c = (int)(t % HH);
  float v = edge_emb[edge_type[e] * HH + c] + tvec[node_batch[srcI[e]] * HH + c];
  ea[t] = v; eah[t] = (_Float16)v;
}

// ------------------------- generic WMMA GEMM -------------------------------
// C[M,Nc] = A[M,K](f16) * Wt[Nc,K]^T(f16) + bias, optional SiLU.
// 256 threads = 8 waves; each wave computes one 16-row tile x NT 16-col tiles,
// reusing a single A fragment across NT accumulators (register blocking).
// M%16==0, Nc%(16*NT)==0, K%32==0.
template <int NT>
__global__ __launch_bounds__(256) void gtd_gemm_wmma(
    const _Float16* __restrict__ A, const _Float16* __restrict__ Bt,
    const float* __restrict__ bias, float* __restrict__ C,
    int M, int K, int Nc, int act) {
  int wave = threadIdx.x >> 5;
  int l    = threadIdx.x & 31;
  int tm   = blockIdx.x * 8 + wave;
  int tn0  = blockIdx.y * NT;
  int koffA = (l >> 4) * 16;
  int koffB = (l >> 4) * 16;
  const _Float16* arow = A  + (size_t)(tm * 16 + (l & 15)) * K;
  const _Float16* brow = Bt + (size_t)(tn0 * 16 + (l & 15)) * K;
  v8f acc[NT];
#pragma unroll
  for (int t = 0; t < NT; ++t) acc[t] = (v8f){};
  for (int kb = 0; kb < K; kb += 32) {
    v8h a0 = *(const v8h*)(arow + kb + koffA);
    v8h a1 = *(const v8h*)(arow + kb + koffA + 8);
    v16h av = __builtin_shufflevector(a0, a1, 0,1,2,3,4,5,6,7,8,9,10,11,12,13,14,15);
#pragma unroll
    for (int t = 0; t < NT; ++t) {
      v16h bv = *(const v16h*)(brow + (size_t)t * 16 * K + kb + koffB);
      acc[t] = __builtin_amdgcn_wmma_f32_16x16x32_f16(false, av, false, bv,
                                                      (short)0, acc[t], false, false);
    }
  }
  int rbase = tm * 16 + (l >> 4) * 8;
#pragma unroll
  for (int t = 0; t < NT; ++t) {
    int col = (tn0 + t) * 16 + (l & 15);
    float bc = bias ? bias[col] : 0.f;
#pragma unroll
    for (int j = 0; j < 8; ++j) {
      float v = acc[t][j] + bc;
      if (act) v = silu_f(v);
      C[(size_t)(rbase + j) * Nc + col] = v;
    }
  }
}

// ------------- edge-MLP layer 1: gathered-A WMMA GEMM (K=768) --------------
// A row m (edge) = [ x[src] | x[dst] | edge_attr | tvec[node_batch[src]] ]
// NT=4 N-tiles per wave: the expensive gathered A fragment is reused 4x.
__global__ __launch_bounds__(256) void gtd_gemm_edge1_wmma(
    const _Float16* __restrict__ xh, const _Float16* __restrict__ eah,
    const _Float16* __restrict__ tvech,
    const int* __restrict__ srcI, const int* __restrict__ dstI,
    const int* __restrict__ nbatch,
    const _Float16* __restrict__ Bt, const float* __restrict__ bias,
    float* __restrict__ C) {
  const int K = 4 * HH, Nc = HH, NT = 4;
  int wave = threadIdx.x >> 5;
  int l    = threadIdx.x & 31;
  int tm   = blockIdx.x * 8 + wave;
  int tn0  = blockIdx.y * NT;
  int m    = tm * 16 + (l & 15);           // edge index this lane owns for A
  int s    = srcI[m], d = dstI[m];
  const _Float16* seg[4] = {
    xh    + (size_t)s * HH,
    xh    + (size_t)d * HH,
    eah   + (size_t)m * HH,
    tvech + (size_t)nbatch[s] * HH
  };
  int koffA = (l >> 4) * 16;
  int koffB = (l >> 4) * 16;
  const _Float16* brow = Bt + (size_t)(tn0 * 16 + (l & 15)) * K;
  v8f acc[NT];
#pragma unroll
  for (int t = 0; t < NT; ++t) acc[t] = (v8f){};
#pragma unroll
  for (int kb = 0; kb < K; kb += 32) {
    const _Float16* base = seg[kb / HH] + (kb % HH);
    v8h a0 = *(const v8h*)(base + koffA);
    v8h a1 = *(const v8h*)(base + koffA + 8);
    v16h av = __builtin_shufflevector(a0, a1, 0,1,2,3,4,5,6,7,8,9,10,11,12,13,14,15);
#pragma unroll
    for (int t = 0; t < NT; ++t) {
      v16h bv = *(const v16h*)(brow + (size_t)t * 16 * K + kb + koffB);
      acc[t] = __builtin_amdgcn_wmma_f32_16x16x32_f16(false, av, false, bv,
                                                      (short)0, acc[t], false, false);
    }
  }
  int rbase = tm * 16 + (l >> 4) * 8;
#pragma unroll
  for (int t = 0; t < NT; ++t) {
    int col = (tn0 + t) * 16 + (l & 15);
    float bc = bias[col];
#pragma unroll
    for (int j = 0; j < 8; ++j) {
      float v = silu_f(acc[t][j] + bc);            // fused SiLU
      C[(size_t)(rbase + j) * Nc + col] = v;
    }
  }
}

// ------------------------- attention passes --------------------------------
__global__ void gtd_attn_init(unsigned* __restrict__ mmax, float* __restrict__ denom,
                              float* __restrict__ agg) {
  long long t = (long long)blockIdx.x * blockDim.x + threadIdx.x;
  if (t >= (long long)NN * HH) return;
  agg[t] = 0.f;
  int c = (int)(t % HH);
  if (c < HEADS) {
    int n = (int)(t / HH);
    mmax[n * HEADS + c]  = 0u;     // 0 < fmap(-inf): identity for atomicMax
    denom[n * HEADS + c] = 0.f;
  }
}

__global__ void gtd_attn_score(const float* __restrict__ q, const float* __restrict__ k,
                               const float* __restrict__ eV,
                               const int* __restrict__ srcI, const int* __restrict__ dstI,
                               float* __restrict__ score, unsigned* __restrict__ mmax) {
  long long t = (long long)blockIdx.x * blockDim.x + threadIdx.x;
  if (t >= (long long)EE * HEADS) return;
  int e = (int)(t >> 2), h = (int)(t & 3);
  int s = srcI[e], d = dstI[e];
  const float* qp = q  + (size_t)d * HH + h * CC;
  const float* kp = k  + (size_t)s * HH + h * CC;
  const float* ep = eV + (size_t)e * HH + h * CC;
  float acc = 0.f;
#pragma unroll 8
  for (int c = 0; c < CC; ++c) acc += qp[c] * (kp[c] + ep[c]);
  acc *= 0.14433756729740643f;      // 1/sqrt(48)
  score[t] = acc;
  atomicMax(mmax + (size_t)d * HEADS + h, fmap(acc));
}

__global__ void gtd_attn_exp(float* __restrict__ score, const unsigned* __restrict__ mmax,
                             const int* __restrict__ dstI, float* __restrict__ denom) {
  long long t = (long long)blockIdx.x * blockDim.x + threadIdx.x;
  if (t >= (long long)EE * HEADS) return;
  int e = (int)(t >> 2), h = (int)(t & 3);
  int d = dstI[e];
  float m  = funmap(mmax[(size_t)d * HEADS + h]);
  float ex = __expf(score[t] - m);
  score[t] = ex;
  atomicAdd(denom + (size_t)d * HEADS + h, ex);
}

__global__ void gtd_attn_agg(const float* __restrict__ v, const float* __restrict__ eV,
                             const float* __restrict__ score, const float* __restrict__ denom,
                             const int* __restrict__ srcI, const int* __restrict__ dstI,
                             float* __restrict__ agg) {
  long long t = (long long)blockIdx.x * blockDim.x + threadIdx.x;
  if (t >= (long long)EE * HH) return;
  int e = (int)(t / HH), c = (int)(t % HH), h = c / CC;
  int s = srcI[e], d = dstI[e];
  float alpha = score[(size_t)e * HEADS + h] / denom[(size_t)d * HEADS + h];
  float msg   = (v[(size_t)s * HH + c] + eV[(size_t)e * HH + c]) * alpha;
  atomicAdd(agg + (size_t)d * HH + c, msg);
}

// ----------------- fused gate + residual + LayerNorm -----------------------
// one block (256 threads) per node
__global__ __launch_bounds__(256) void gtd_fuse_ln(
    float* __restrict__ x, _Float16* __restrict__ xh,
    const float* __restrict__ agg, const float* __restrict__ xr,
    const float* __restrict__ wbeta, const float* __restrict__ g,
    const float* __restrict__ bvec) {
  __shared__ float red[256];
  int n = blockIdx.x, j = threadIdx.x;
  float o = 0.f, r = 0.f, partial = 0.f;
  if (j < HH) {
    o = agg[(size_t)n * HH + j];
    r = xr[(size_t)n * HH + j];
    partial = o * wbeta[j] + r * wbeta[HH + j] + (o - r) * wbeta[2 * HH + j];
  }
  red[j] = partial;
  __syncthreads();
  for (int s = 128; s > 0; s >>= 1) { if (j < s) red[j] += red[j + s]; __syncthreads(); }
  float beta = 1.f / (1.f + __expf(-red[0]));
  __syncthreads();
  float y = 0.f;
  if (j < HH) {
    float hgate = beta * r + (1.f - beta) * o;
    y = x[(size_t)n * HH + j] + hgate;
  }
  red[j] = (j < HH) ? y : 0.f;
  __syncthreads();
  for (int s = 128; s > 0; s >>= 1) { if (j < s) red[j] += red[j + s]; __syncthreads(); }
  float mu = red[0] / (float)HH;
  __syncthreads();
  float dy = y - mu;
  red[j] = (j < HH) ? dy * dy : 0.f;
  __syncthreads();
  for (int s = 128; s > 0; s >>= 1) { if (j < s) red[j] += red[j + s]; __syncthreads(); }
  float var = red[0] / (float)HH;
  if (j < HH) {
    float out = dy * rsqrtf(var + 1e-5f) * g[j] + bvec[j];
    x[(size_t)n * HH + j]  = out;
    xh[(size_t)n * HH + j] = (_Float16)out;
  }
}

// ------------------------- edge-MLP layer 2 --------------------------------
// edge_logits[E,8] = hidden[E,192] @ w2[192,8] + b2
__global__ void gtd_edge_out(const float* __restrict__ hidden,
                             const float* __restrict__ w2, const float* __restrict__ b2,
                             float* __restrict__ out) {
  long long t = (long long)blockIdx.x * blockDim.x + threadIdx.x;
  if (t >= (long long)EE * VE_) return;
  int e = (int)(t >> 3), o = (int)(t & 7);
  float acc = b2[o];
  const float* hp = hidden + (size_t)e * HH;
#pragma unroll 8
  for (int i = 0; i < HH; ++i) acc += hp[i] * w2[i * VE_ + o];
  out[t] = acc;
}

// ---------------------------------------------------------------------------

extern "C" void kernel_launch(void* const* d_in, const int* in_sizes, int n_in,
                              void* d_out, int out_size, void* d_ws, size_t ws_size,
                              hipStream_t stream) {
  // ---- inputs (setup_inputs order) ----
  const float* node_emb   = (const float*)d_in[0];
  const float* edge_emb   = (const float*)d_in[1];
  const float* time_w1    = (const float*)d_in[2];
  const float* time_b1    = (const float*)d_in[3];
  const float* time_w2    = (const float*)d_in[4];
  const float* time_b2    = (const float*)d_in[5];
  const float* mask_emb   = (const float*)d_in[6];
  const float* wq         = (const float*)d_in[7];
  const float* bq         = (const float*)d_in[8];
  const float* wk         = (const float*)d_in[9];
  const float* bk         = (const float*)d_in[10];
  const float* wv         = (const float*)d_in[11];
  const float* bv         = (const float*)d_in[12];
  const float* we         = (const float*)d_in[13];
  const float* wskip      = (const float*)d_in[14];
  const float* bskip      = (const float*)d_in[15];
  const float* wbeta      = (const float*)d_in[16];
  const float* ln_g       = (const float*)d_in[17];
  const float* ln_b       = (const float*)d_in[18];
  const float* node_out_w = (const float*)d_in[19];
  const float* node_out_b = (const float*)d_in[20];
  const float* edge_w1    = (const float*)d_in[21];
  const float* edge_b1    = (const float*)d_in[22];
  const float* edge_w2    = (const float*)d_in[23];
  const float* edge_b2    = (const float*)d_in[24];
  const int*   node_type  = (const int*)d_in[25];
  const int*   edge_type  = (const int*)d_in[26];
  const int*   edge_index = (const int*)d_in[27];
  const int*   edit_mask  = (const int*)d_in[28];
  const int*   node_batch = (const int*)d_in[29];
  const int*   t_graph    = (const int*)d_in[30];

  const int* srcI = edge_index;
  const int* dstI = edge_index + EE;

  // ---- workspace carve-up (256B aligned) ----
  char* wp = (char*)d_ws;
  auto alloc = [&](size_t bytes) -> void* {
    void* r = (void*)wp;
    wp += (bytes + 255) & ~(size_t)255;
    return r;
  };
  const size_t HW = (size_t)HH * HH;  // 36864
  _Float16* wt_layer = (_Float16*)alloc(LL * 5 * HW * sizeof(_Float16)); // [l][q,k,v,e,skip]
  _Float16* wt_nout  = (_Float16*)alloc((size_t)HH * VN_ * sizeof(_Float16));
  _Float16* wt_e1    = (_Float16*)alloc((size_t)4 * HH * HH * sizeof(_Float16));
  float*    tvec     = (float*)alloc((size_t)BB * HH * sizeof(float));
  _Float16* tvech    = (_Float16*)alloc((size_t)BB * HH * sizeof(_Float16));
  float*    x        = (float*)alloc((size_t)NN * HH * sizeof(float));
  _Float16* xh       = (_Float16*)alloc((size_t)NN * HH * sizeof(_Float16));
  float*    ea       = (float*)alloc((size_t)EE * HH * sizeof(float));
  _Float16* eah      = (_Float16*)alloc((size_t)EE * HH * sizeof(_Float16));
  float*    qb       = (float*)alloc((size_t)NN * HH * sizeof(float));
  float*    kb       = (float*)alloc((size_t)NN * HH * sizeof(float));
  float*    vb       = (float*)alloc((size_t)NN * HH * sizeof(float));
  float*    xr       = (float*)alloc((size_t)NN * HH * sizeof(float));
  float*    eV       = (float*)alloc((size_t)EE * HH * sizeof(float));
  float*    score    = (float*)alloc((size_t)EE * HEADS * sizeof(float));
  unsigned* mmax     = (unsigned*)alloc((size_t)NN * HEADS * sizeof(unsigned));
  float*    denom    = (float*)alloc((size_t)NN * HEADS * sizeof(float));
  float*    agg      = (float*)alloc((size_t)NN * HH * sizeof(float));
  float*    hidden   = (float*)alloc((size_t)EE * HH * sizeof(float));

  float* node_logits = (float*)d_out;                       // [N, 32]
  float* edge_logits = (float*)d_out + (size_t)NN * VN_;    // [E, 8]

  const int TB = 256;
  const int convHH = (int)((HW + TB - 1) / TB);

  // ---- weight conversion (f32 -> transposed f16) ----
  for (int l = 0; l < LL; ++l) {
    gtd_convT<<<convHH, TB, 0, stream>>>(wq    + (size_t)l * HW, wt_layer + ((size_t)l * 5 + 0) * HW, HH, HH);
    gtd_convT<<<convHH, TB, 0, stream>>>(wk    + (size_t)l * HW, wt_layer + ((size_t)l * 5 + 1) * HW, HH, HH);
    gtd_convT<<<convHH, TB, 0, stream>>>(wv    + (size_t)l * HW, wt_layer + ((size_t)l * 5 + 2) * HW, HH, HH);
    gtd_convT<<<convHH, TB, 0, stream>>>(we    + (size_t)l * HW, wt_layer + ((size_t)l * 5 + 3) * HW, HH, HH);
    gtd_convT<<<convHH, TB, 0, stream>>>(wskip + (size_t)l * HW, wt_layer + ((size_t)l * 5 + 4) * HW, HH, HH);
  }
  gtd_convT<<<(HH * VN_ + TB - 1) / TB, TB, 0, stream>>>(node_out_w, wt_nout, HH, VN_);
  gtd_convT<<<(4 * HH * HH + TB - 1) / TB, TB, 0, stream>>>(edge_w1, wt_e1, 4 * HH, HH);

  // ---- time embedding ----
  gtd_time_embed<<<BB, HH, 0, stream>>>(t_graph, time_w1, time_b1, time_w2, time_b2,
                                        tvec, tvech);

  // ---- initial node / edge features ----
  {
    long long nt = (long long)NN * HH;
    gtd_init_x<<<(unsigned)((nt + TB - 1) / TB), TB, 0, stream>>>(
        node_emb, mask_emb, tvec, node_type, edit_mask, node_batch, x, xh);
    long long et = (long long)EE * HH;
    gtd_init_edge<<<(unsigned)((et + TB - 1) / TB), TB, 0, stream>>>(
        edge_emb, tvec, edge_type, srcI, node_batch, ea, eah);
  }

  dim3 gN(NN / 128, HH / 64);   // node GEMMs: 125 x 3, NT=4
  dim3 gE(EE / 128, HH / 64);   // edge GEMMs: 2000 x 3, NT=4

  for (int l = 0; l < LL; ++l) {
    const _Float16* wtq = wt_layer + ((size_t)l * 5 + 0) * HW;
    const _Float16* wtk = wt_layer + ((size_t)l * 5 + 1) * HW;
    const _Float16* wtv = wt_layer + ((size_t)l * 5 + 2) * HW;
    const _Float16* wte = wt_layer + ((size_t)l * 5 + 3) * HW;
    const _Float16* wts = wt_layer + ((size_t)l * 5 + 4) * HW;

    gtd_gemm_wmma<4><<<gN, TB, 0, stream>>>(xh, wtq, bq + (size_t)l * HH, qb, NN, HH, HH, 0);
    gtd_gemm_wmma<4><<<gN, TB, 0, stream>>>(xh, wtk, bk + (size_t)l * HH, kb, NN, HH, HH, 0);
    gtd_gemm_wmma<4><<<gN, TB, 0, stream>>>(xh, wtv, bv + (size_t)l * HH, vb, NN, HH, HH, 0);
    gtd_gemm_wmma<4><<<gN, TB, 0, stream>>>(xh, wts, bskip + (size_t)l * HH, xr, NN, HH, HH, 0);
    gtd_gemm_wmma<4><<<gE, TB, 0, stream>>>(eah, wte, nullptr, eV, EE, HH, HH, 0);

    long long nt = (long long)NN * HH;
    gtd_attn_init<<<(unsigned)((nt + TB - 1) / TB), TB, 0, stream>>>(mmax, denom, agg);

    long long eh = (long long)EE * HEADS;
    gtd_attn_score<<<(unsigned)((eh + TB - 1) / TB), TB, 0, stream>>>(
        qb, kb, eV, srcI, dstI, score, mmax);
    gtd_attn_exp<<<(unsigned)((eh + TB - 1) / TB), TB, 0, stream>>>(
        score, mmax, dstI, denom);

    long long ec = (long long)EE * HH;
    gtd_attn_agg<<<(unsigned)((ec + TB - 1) / TB), TB, 0, stream>>>(
        vb, eV, score, denom, srcI, dstI, agg);

    gtd_fuse_ln<<<NN, TB, 0, stream>>>(x, xh, agg, xr,
                                       wbeta + (size_t)l * 3 * HH,
                                       ln_g + (size_t)l * HH,
                                       ln_b + (size_t)l * HH);
  }

  // ---- node logits: [N,192] @ [192,32]  (NT=2, grid.y=1) ----
  dim3 gNO(NN / 128, 1);
  gtd_gemm_wmma<2><<<gNO, TB, 0, stream>>>(xh, wt_nout, node_out_b, node_logits,
                                           NN, HH, VN_, 0);

  // ---- edge MLP: gathered [E,768] @ [768,192] (+SiLU), then [192,8] ----
  gtd_gemm_edge1_wmma<<<gE, TB, 0, stream>>>(xh, eah, tvech, srcI, dstI, node_batch,
                                             wt_e1, edge_b1, hidden);
  long long eo = (long long)EE * VE_;
  gtd_edge_out<<<(unsigned)((eo + TB - 1) / TB), TB, 0, stream>>>(
      hidden, edge_w2, edge_b2, edge_logits);
}
